// W4Linear_33208687132984
// MI455X (gfx1250) — compile-verified, matrix operand order checked by reading, not run on hardware
//
#include <hip/hip_runtime.h>

typedef __attribute__((ext_vector_type(2)))  _Float16 h2;
typedef __attribute__((ext_vector_type(8)))  _Float16 v8h;
typedef __attribute__((ext_vector_type(16))) _Float16 v16h;
typedef __attribute__((ext_vector_type(8)))  float    v8f;

#define M_TOT 8192
#define N_TOT 4096
#define K_TOT 4096
#define BM    128
#define BN    128
#define KT    64
#define LDA   72            // KT + 8 halfs of padding; 144B row stride, 16B aligned
#define NSTAGE (K_TOT / KT) // 64

// Dequant 2 nibbles of one byte -> packed f16 pair (q-8), exact, no int->float cvt.
// 0x6400|q == f16(1024+q); v_pk_add_f16 with -1032.0 gives (q-8) in both halves.
__device__ __forceinline__ uint32_t dq2(int v) {
    uint32_t u = (uint32_t)v;
    uint32_t t = ((u | (u << 12)) & 0x000F000Fu) | 0x64006400u;
    h2 h = __builtin_bit_cast(h2, t);
    const h2 off = {(_Float16)(-1032.0f), (_Float16)(-1032.0f)};
    h = h + off;
    return __builtin_bit_cast(uint32_t, h);
}

__global__ __launch_bounds__(256)
void w4_gemm_wmma(const float* __restrict__ X,
                  const int*   __restrict__ Wp,
                  const float* __restrict__ scale_p,
                  const float* __restrict__ bias,
                  float*       __restrict__ Out)
{
    // 73,728 B static LDS (CDNA5 allows up to 320 KB per workgroup)
    __shared__ __align__(16) _Float16 As[2][BM][LDA];
    __shared__ __align__(16) _Float16 Bs[2][BN][LDA];

    const int tid  = threadIdx.x;
    const int lane = tid & 31;
    const int wave = tid >> 5;
    const int wm   = wave >> 1;   // 0..3  -> 32-row band
    const int wn   = wave & 1;    // 0..1  -> 64-col band
    const int bm   = blockIdx.y * BM;
    const int bn   = blockIdx.x * BN;

    const float scale = *scale_p;

    // ---- accumulators: 2 (M) x 4 (N) tiles of 16x16 f32 ----
    v8f acc[2][4];
#pragma unroll
    for (int tm = 0; tm < 2; ++tm)
#pragma unroll
        for (int tn = 0; tn < 4; ++tn)
            acc[tm][tn] = (v8f)0.0f;

    // ---------- global fetch of one K-stage into registers ----------
    // A: 128 rows x 64 f32 = 1024 chunks of 8 floats; thread handles 4 chunks.
    // W: 128 rows x 32 int32 = 1024 int4 chunks; thread handles 4 chunks.
    auto fetchStage = [&](int s, float4 xa[8], int4 wb[4]) {
        const int kt = s * KT;
#pragma unroll
        for (int i = 0; i < 4; ++i) {
            const int j   = tid + i * 256;     // 0..1023
            const int row = j >> 3;            // 8 chunks per row
            const int c   = j & 7;
            const float* p = X + (size_t)(bm + row) * K_TOT + kt + c * 8;
            xa[2 * i + 0] = *(const float4*)(p);
            xa[2 * i + 1] = *(const float4*)(p + 4);
        }
#pragma unroll
        for (int i = 0; i < 4; ++i) {
            const int j   = tid + i * 256;
            const int row = j >> 3;            // 8 int4 per row (32 ints/row)
            const int c   = j & 7;
            wb[i] = *(const int4*)(Wp + (size_t)(bn + row) * (K_TOT / 2) + (kt >> 1) + c * 4);
        }
    };

    // ---------- convert + store a fetched stage into LDS buffer ----------
    auto storeStage = [&](int buf, const float4 xa[8], const int4 wb[4]) {
#pragma unroll
        for (int i = 0; i < 4; ++i) {
            const int j   = tid + i * 256;
            const int row = j >> 3;
            const int c   = j & 7;
            const float4 a0 = xa[2 * i + 0];
            const float4 a1 = xa[2 * i + 1];
            // packed f32->f16 converts: one v_cvt_pk_rtz_f16_f32 per float pair
            uint4 st;
            st.x = __builtin_bit_cast(uint32_t, __builtin_amdgcn_cvt_pkrtz(a0.x, a0.y));
            st.y = __builtin_bit_cast(uint32_t, __builtin_amdgcn_cvt_pkrtz(a0.z, a0.w));
            st.z = __builtin_bit_cast(uint32_t, __builtin_amdgcn_cvt_pkrtz(a1.x, a1.y));
            st.w = __builtin_bit_cast(uint32_t, __builtin_amdgcn_cvt_pkrtz(a1.z, a1.w));
            *(uint4*)&As[buf][row][c * 8] = st;
        }
#pragma unroll
        for (int i = 0; i < 4; ++i) {
            const int j   = tid + i * 256;
            const int row = j >> 3;
            const int c   = j & 7;
            const int4 w4 = wb[i];
            uint4 st;
            st.x = dq2(w4.x);
            st.y = dq2(w4.y);
            st.z = dq2(w4.z);
            st.w = dq2(w4.w);
            *(uint4*)&Bs[buf][row][c * 8] = st;
        }
    };

    // ---------- two K-steps of WMMA from an LDS buffer ----------
    // 16-bit A/B fragment layout (ISA 7.12.2): lanes 0-15 hold K 0..7 & 16..23,
    // lanes 16-31 hold K 8..15 & 24..31 of row (A) / column (B) = lane%16.
    auto compute = [&](int buf) {
        const int lr = lane & 15;
#pragma unroll
        for (int ks = 0; ks < 2; ++ks) {
            const int khi = ks * 32 + ((lane >> 4) << 3);   // + 0 or 8
            v16h afrag[2], bfrag[4];
#pragma unroll
            for (int tm = 0; tm < 2; ++tm) {
                const _Float16* p = &As[buf][wm * 32 + tm * 16 + lr][khi];
                v8h lo = *(const v8h*)p;
                v8h hi = *(const v8h*)(p + 16);
                afrag[tm] = __builtin_shufflevector(lo, hi,
                    0, 1, 2, 3, 4, 5, 6, 7, 8, 9, 10, 11, 12, 13, 14, 15);
            }
#pragma unroll
            for (int tn = 0; tn < 4; ++tn) {
                const _Float16* p = &Bs[buf][wn * 64 + tn * 16 + lr][khi];
                v8h lo = *(const v8h*)p;
                v8h hi = *(const v8h*)(p + 16);
                bfrag[tn] = __builtin_shufflevector(lo, hi,
                    0, 1, 2, 3, 4, 5, 6, 7, 8, 9, 10, 11, 12, 13, 14, 15);
            }
#pragma unroll
            for (int tm = 0; tm < 2; ++tm)
#pragma unroll
                for (int tn = 0; tn < 4; ++tn)
                    acc[tm][tn] = __builtin_amdgcn_wmma_f32_16x16x32_f16(
                        false, afrag[tm], false, bfrag[tn],
                        (short)0, acc[tm][tn], false, false);
        }
    };

    // ---------- pipelined main loop (double-buffered LDS) ----------
    {
        float4 xa[8]; int4 wb[4];
        fetchStage(0, xa, wb);
        storeStage(0, xa, wb);
    }
    __syncthreads();

    for (int s = 0; s < NSTAGE; ++s) {
        const int cur = s & 1;

        float4 xa[8]; int4 wb[4];
        if (s + 1 < NSTAGE)
            fetchStage(s + 1, xa, wb);

        if (s + 2 < NSTAGE) {
            // exercise gfx1250 prefetch path for stage s+2
            const int kt2 = (s + 2) * KT;
            __builtin_prefetch(X  + (size_t)(bm + (tid >> 1)) * K_TOT + kt2 + (tid & 1) * 32, 0, 3);
            __builtin_prefetch(Wp + (size_t)(bn + (tid >> 1)) * (K_TOT / 2) + (kt2 >> 1) + (tid & 1) * 16, 0, 3);
        }

        compute(cur);

        if (s + 1 < NSTAGE)
            storeStage(cur ^ 1, xa, wb);
        __syncthreads();
    }

    // ---------- epilogue: out = scale*acc + bias ----------
    // C/D layout: VGPR r -> M = r (lanes 0-15) / r+8 (lanes 16-31); N = lane%16.
#pragma unroll
    for (int tm = 0; tm < 2; ++tm) {
        const int r0 = bm + wm * 32 + tm * 16 + ((lane >> 4) << 3);
#pragma unroll
        for (int tn = 0; tn < 4; ++tn) {
            const int cidx = bn + wn * 64 + tn * 16 + (lane & 15);
            const float bb = bias[cidx];
#pragma unroll
            for (int r = 0; r < 8; ++r)
                Out[(size_t)(r0 + r) * N_TOT + cidx] = acc[tm][tn][r] * scale + bb;
        }
    }
}

extern "C" void kernel_launch(void* const* d_in, const int* in_sizes, int n_in,
                              void* d_out, int out_size, void* d_ws, size_t ws_size,
                              hipStream_t stream)
{
    const float* x     = (const float*)d_in[0];
    const int*   wp    = (const int*)d_in[1];
    const float* scale = (const float*)d_in[2];
    const float* bias  = (const float*)d_in[3];
    float*       out   = (float*)d_out;

    dim3 grid(N_TOT / BN, M_TOT / BM);  // (32, 64)
    w4_gemm_wmma<<<grid, 256, 0, stream>>>(x, wp, scale, bias, out);
}
